// DeepAR_65687229825238
// MI455X (gfx1250) — matrix operand order repaired and checked by reading, not compiled, not run on hardware
//
#include <hip/hip_runtime.h>
#include <math.h>

// ---------------------------------------------------------------------------
// DeepAR decode on MI455X (gfx1250, wave32).
//
// Key insight: the reference "LSTM" uses zero state every step -> pure
// feed-forward per timestep. Only steps t=1023..1150 (128 steps) affect the
// output, chained through the scalar likelihood feedback. Each step is
// three matvecs (4096x64, 4096x1024, 4096x1024) -> L2-resident, bandwidth
// bound. We run them on WMMA f16 tiles with a persistent 16-WG grid.
// ---------------------------------------------------------------------------

typedef _Float16 v16h __attribute__((ext_vector_type(16)));
typedef _Float16 v8h  __attribute__((ext_vector_type(8)));
typedef float    v8f  __attribute__((ext_vector_type(8)));

#define H4     4096      // 4*H
#define HH     1024      // H
#define K0DIM  64        // F + E
#define NWG    16
#define BLOCK  512       // 16 waves of 32
#define NWAVES (NWG * (BLOCK/32))   // 256 waves == 256 row tiles of 16
#define HORZ   128
#define TWO_PI 6.28318530717958647692f

// ---------------- grid barrier (monotonic counter, graph-replay safe) ------
__device__ __forceinline__ void gbar(unsigned int* cnt, unsigned int& epoch) {
    __syncthreads();
    if (threadIdx.x == 0) {
        __threadfence();
        __hip_atomic_fetch_add(cnt, 1u, __ATOMIC_RELEASE, __HIP_MEMORY_SCOPE_AGENT);
        epoch += NWG;
        while (__hip_atomic_load(cnt, __ATOMIC_ACQUIRE, __HIP_MEMORY_SCOPE_AGENT) < epoch) {
            __builtin_amdgcn_s_sleep(4);
        }
        __threadfence();
    }
    __syncthreads();
}

__device__ __forceinline__ float sigf(float x) { return 1.0f / (1.0f + __expf(-x)); }

// ---------------- WMMA matvec: 16 output rows per wave ---------------------
// A tile = 16 rows x 32 K of f16 weights (row-major).  Per the CDNA5 ISA A
// layout (16-bit A 16x32): lane = M + 16*halfsel; lanes 0-15 hold K 0-7 and
// 16-23, lanes 16-31 hold K 8-15 and 24-31  -> two contiguous 16B loads/lane.
// B = input vector x broadcast into all 16 columns, same K order as A:
// lanes 0-15 hold K 0-7 and 16-23, lanes 16-31 hold K 8-15 and 24-31 -> two
// 16B LDS reads. Every column of D then equals W_tile @ x.
__device__ __forceinline__ void matvec_wmma(const _Float16* __restrict__ W, int K,
                                            const _Float16* __restrict__ xlds,
                                            const float* __restrict__ bsum,
                                            float* __restrict__ gout,
                                            int R, int lane) {
    const int m  = lane & 15;
    const int hi = lane >> 4;
    const _Float16* wrow = W + (size_t)(R + m) * (size_t)K;
    v8f acc = {};
#pragma unroll 4
    for (int k0 = 0; k0 < K; k0 += 32) {
        __builtin_prefetch(wrow + k0 + 128, 0, 1);   // global_prefetch_b8
        v8h alo = *(const v8h*)(wrow + k0      + 8 * hi);
        v8h ahi = *(const v8h*)(wrow + k0 + 16 + 8 * hi);
        v16h a = __builtin_shufflevector(alo, ahi, 0,1,2,3,4,5,6,7,8,9,10,11,12,13,14,15);
        v8h blo = *(const v8h*)(xlds + k0      + 8 * hi);
        v8h bhi = *(const v8h*)(xlds + k0 + 16 + 8 * hi);
        v16h b = __builtin_shufflevector(blo, bhi, 0,1,2,3,4,5,6,7,8,9,10,11,12,13,14,15);
        acc = __builtin_amdgcn_wmma_f32_16x16x32_f16(
                  false, a, false, b, (short)0, acc, false, false);
    }
    if (m == 0) {                         // lanes 0 & 16: rows R+8*hi .. +7
        const int rbase = R + 8 * hi;
#pragma unroll
        for (int r = 0; r < 8; ++r)
            gout[rbase + r] = acc[r] + bsum[rbase + r];
    }
}

// gate nonlinearity: i,g,o gates -> h, written to LDS as f16 (next layer's B)
__device__ __forceinline__ void act_to_lds(const float* __restrict__ g,
                                           _Float16* __restrict__ hlds, int tid) {
#pragma unroll
    for (int r = 0; r < 2; ++r) {
        const int i = tid * 2 + r;
        const float ig = g[i], gg = g[i + 2048], og = g[i + 3072];
        const float c  = sigf(ig) * tanhf(gg);
        const float h  = sigf(og) * tanhf(c);
        hlds[i] = (_Float16)h;
    }
    __syncthreads();
}

// ---------------- one-time weight conversion + barrier reset ---------------
__global__ void deepar_convert(const float* __restrict__ w_ih0,
                               const float* __restrict__ b_ih0,
                               const float* __restrict__ b_hh0,
                               const float* __restrict__ w_ih_r,
                               const float* __restrict__ b_ih_r,
                               const float* __restrict__ b_hh_r,
                               const float* __restrict__ y,
                               _Float16* __restrict__ w0,
                               _Float16* __restrict__ w1,
                               _Float16* __restrict__ w2,
                               float* __restrict__ bsum0,
                               float* __restrict__ bsum1,
                               float* __restrict__ bsum2,
                               float* __restrict__ ycarry,
                               unsigned int* __restrict__ cnt) {
    const size_t i  = (size_t)blockIdx.x * blockDim.x + threadIdx.x;
    const size_t NR = (size_t)H4 * HH;           // 4,194,304
    if (i < NR) {
        w1[i] = (_Float16)w_ih_r[i];
        w2[i] = (_Float16)w_ih_r[NR + i];
    }
    if (i < (size_t)H4 * K0DIM) w0[i] = (_Float16)w_ih0[i];
    if (i < H4) {
        bsum0[i] = b_ih0[i] + b_hh0[i];
        bsum1[i] = b_ih_r[i] + b_hh_r[i];
        bsum2[i] = b_ih_r[H4 + i] + b_hh_r[H4 + i];
    }
    if (i == 0) {
        *ycarry = y[1023];   // step t=1023 is observed: yin = y[1023]
        *cnt    = 0u;        // reset grid barrier every launch (replay-safe)
    }
}

// ---------------- persistent decode kernel ---------------------------------
__global__ __launch_bounds__(BLOCK)
void deepar_decode(const float* __restrict__ X,  const float* __restrict__ Xf,
                   const float* __restrict__ We, const float* __restrict__ be,
                   const float* __restrict__ Wmu, const float* __restrict__ bmu,
                   const float* __restrict__ Wsig, const float* __restrict__ bsig,
                   const _Float16* __restrict__ w0, const _Float16* __restrict__ w1,
                   const _Float16* __restrict__ w2,
                   const float* __restrict__ bsum0, const float* __restrict__ bsum1,
                   const float* __restrict__ bsum2,
                   float* __restrict__ g0, float* __restrict__ g1,
                   float* __restrict__ g2,
                   float* __restrict__ ycarry, unsigned int* __restrict__ cnt,
                   float* __restrict__ out) {
    __shared__ __align__(32) _Float16 xin[K0DIM];
    __shared__ __align__(32) _Float16 hbuf[HH];
    __shared__ float hf[HH];
    __shared__ float redm[BLOCK];
    __shared__ float reds[BLOCK];

    const int tid  = threadIdx.x;
    const int lane = tid & 31;
    const int wave = tid >> 5;
    const int R    = (blockIdx.x * (BLOCK / 32) + wave) * 16;  // 16-row tile
    unsigned int epoch = 0;

    for (int s = 0; s < HORZ; ++s) {
        // ---- build the 64-wide layer-0 input in LDS (redundant per WG) ----
        const float yin = *(volatile const float*)ycarry;   // step-end barrier ordered
        if (tid < K0DIM) {
            float v;
            if (tid < 32) v = (s == 0) ? X[1023 * 32 + tid] : Xf[(s - 1) * 32 + tid];
            else          v = We[tid - 32] * yin + be[tid - 32];
            xin[tid] = (_Float16)v;
        }
        __syncthreads();

        // ---- layer 0: 4096 x 64 ----
        matvec_wmma(w0, K0DIM, xin, bsum0, g0, R, lane);
        gbar(cnt, epoch);

        // ---- layer 1: 4096 x 1024 ----
        act_to_lds(g0, hbuf, tid);
        matvec_wmma(w1, HH, hbuf, bsum1, g1, R, lane);
        gbar(cnt, epoch);

        // ---- layer 2: 4096 x 1024 ----
        act_to_lds(g1, hbuf, tid);
        matvec_wmma(w2, HH, hbuf, bsum2, g2, R, lane);
        gbar(cnt, epoch);

        // ---- head (WG0 only): mu / sigma / likelihood ----
        if (blockIdx.x == 0) {
#pragma unroll
            for (int r = 0; r < 2; ++r) {
                const int i = tid * 2 + r;
                const float ig = g2[i], gg = g2[i + 2048], og = g2[i + 3072];
                const float c  = sigf(ig) * tanhf(gg);
                hf[i] = sigf(og) * tanhf(c);
            }
            __syncthreads();
            const int i0 = tid * 2;
            redm[tid] = Wmu[i0]  * hf[i0] + Wmu[i0 + 1]  * hf[i0 + 1];
            reds[tid] = Wsig[i0] * hf[i0] + Wsig[i0 + 1] * hf[i0 + 1];
            __syncthreads();
            for (int st = BLOCK / 2; st > 0; st >>= 1) {
                if (tid < st) { redm[tid] += redm[tid + st]; reds[tid] += reds[tid + st]; }
                __syncthreads();
            }
            if (tid == 0) {
                const float mu = redm[0] + bmu[0];
                const float tv = reds[0] + bsig[0];
                const float sp = (tv > 20.0f) ? tv : log1pf(__expf(tv));
                const float sg = sp + 1e-6f;
                const float d  = yin - mu;
                const float lik = rsqrtf(TWO_PI * sg * sg) * __expf(-d * d / (2.0f * sg * sg));
                out[s] = lik;
                *ycarry = lik;          // feedback for step s+1
            }
        }
        gbar(cnt, epoch);   // publishes ycarry before next step's xin build
    }
}

// ---------------------------------------------------------------------------
extern "C" void kernel_launch(void* const* d_in, const int* in_sizes, int n_in,
                              void* d_out, int out_size, void* d_ws, size_t ws_size,
                              hipStream_t stream) {
    const float* X      = (const float*)d_in[0];
    const float* y      = (const float*)d_in[1];
    const float* Xf     = (const float*)d_in[2];
    const float* We     = (const float*)d_in[3];
    const float* be     = (const float*)d_in[4];
    const float* w_ih0  = (const float*)d_in[5];
    const float* b_ih0  = (const float*)d_in[6];
    const float* b_hh0  = (const float*)d_in[7];
    const float* w_ih_r = (const float*)d_in[8];
    const float* b_ih_r = (const float*)d_in[9];
    const float* b_hh_r = (const float*)d_in[10];
    const float* Wmu    = (const float*)d_in[11];
    const float* bmu    = (const float*)d_in[12];
    const float* Wsig   = (const float*)d_in[13];
    const float* bsig   = (const float*)d_in[14];
    float* out = (float*)d_out;

    // workspace layout (~17.5 MB; ws assumed large enough)
    char* ws = (char*)d_ws;
    size_t off = 0;
    auto carve = [&](size_t bytes) { void* p = ws + off; off = (off + bytes + 255) & ~(size_t)255; return p; };
    _Float16* w0   = (_Float16*)carve((size_t)H4 * K0DIM * 2);
    _Float16* w1   = (_Float16*)carve((size_t)H4 * HH * 2);
    _Float16* w2   = (_Float16*)carve((size_t)H4 * HH * 2);
    float* bsum0   = (float*)carve(H4 * 4);
    float* bsum1   = (float*)carve(H4 * 4);
    float* bsum2   = (float*)carve(H4 * 4);
    float* g0      = (float*)carve(H4 * 4);
    float* g1      = (float*)carve(H4 * 4);
    float* g2      = (float*)carve(H4 * 4);
    float* ycarry  = (float*)carve(256);
    unsigned int* cnt = (unsigned int*)carve(256);
    (void)in_sizes; (void)n_in; (void)out_size; (void)ws_size;

    // 1) one-time f16 conversion + bias fusion + barrier/feedback init
    const size_t NR = (size_t)H4 * HH;                 // 4,194,304
    deepar_convert<<<(unsigned)(NR / 256), 256, 0, stream>>>(
        w_ih0, b_ih0, b_hh0, w_ih_r, b_ih_r, b_hh_r, y,
        w0, w1, w2, bsum0, bsum1, bsum2, ycarry, cnt);

    // 2) persistent 16-WG cooperative decode (128 chained steps)
    deepar_decode<<<NWG, BLOCK, 0, stream>>>(
        X, Xf, We, be, Wmu, bmu, Wsig, bsig,
        w0, w1, w2, bsum0, bsum1, bsum2,
        g0, g1, g2, ycarry, cnt, out);
}
